// LinearAttention_61847529062805
// MI455X (gfx1250) — matrix operand order repaired and checked by reading, not compiled, not run on hardware
//
#include <hip/hip_runtime.h>

#define DIM    1024
#define NHEADS 16
#define HDIM   64
#define BATCH  2
#define SEQ    8192
#define NROWS  (BATCH * SEQ)

#define K1_CH    4                       // N-chunks per (b,h) in kv_partial
#define K1_ROWS  (SEQ / K1_CH)           // 2048 rows per block
#define K1_WROWS (K1_ROWS / 8)           // 256 rows per wave
#define K1_ITERS (K1_WROWS / 32)         // 8 WMMA K-steps per wave

#define K2_BLKS  8                       // blocks per (b,h) in attn_norm
#define K2_ROWS  (SEQ / K2_BLKS)         // 1024 rows per block

#if defined(__has_builtin)
#  if __has_builtin(__builtin_amdgcn_global_load_async_to_lds_b128)
#    define HAVE_ASYNC_LDS 1
#  endif
#endif
#ifndef HAVE_ASYNC_LDS
#  define HAVE_ASYNC_LDS 0
#endif

typedef __attribute__((ext_vector_type(16))) _Float16 v16h;
typedef __attribute__((ext_vector_type(8)))  _Float16 v8h;
typedef __attribute__((ext_vector_type(8)))  float    v8f;
typedef __attribute__((ext_vector_type(4)))  int      v4i;

// Pointer-to-AS-qualified-v4i types for the async global->LDS builtin
// (signature expects V4i addrspace(1)* and V4i addrspace(3)*).
typedef __attribute__((address_space(1))) v4i* gas_v4i_ptr;
typedef __attribute__((address_space(3))) v4i* lds_v4i_ptr;

static __device__ __forceinline__ v8f wmma_f16(v16h a, v16h b, v8f c) {
    // D = A(16x32 f16) * B(32x16 f16) + C(16x16 f32)
    return __builtin_amdgcn_wmma_f32_16x16x32_f16(
        /*neg_a=*/false, a, /*neg_b=*/false, b,
        /*c_mod=*/(short)0, c, /*reuse_a=*/false, /*reuse_b=*/false);
}

// A-fragment element i (i = 2*vgpr + pair) maps to K = 16*(i>>3) + 8*half + (i&7)
static __device__ __forceinline__ int a_kpos(int i, int half) {
    return 16 * (i >> 3) + 8 * half + (i & 7);
}

// ---------------------------------------------------------------------------
// Kernel 0: pre-permute W_out^T into per-lane WMMA B-fragment layout.
// wfrag index = ((jtile*32 + kt)*32 + lane)*16 + i
//   element  = W_out[col][K],  col = jtile*16 + (lane&15),
//              K = kt*32 + 16*(i>>3) + 8*(lane>>4) + (i&7)
// ---------------------------------------------------------------------------
__global__ __launch_bounds__(256) void wfrag_kernel(const float* __restrict__ W,
                                                    _Float16* __restrict__ wfrag) {
    const int t     = blockIdx.x * 256 + threadIdx.x;   // 0 .. 2^20-1
    const int i     = t & 15;
    const int lane  = (t >> 4) & 31;
    const int kt    = (t >> 9) & 31;
    const int jtile = t >> 14;
    const int half  = lane >> 4;
    const int m     = lane & 15;
    const int K     = kt * 32 + a_kpos(i, half);
    const int col   = jtile * 16 + m;
    wfrag[t] = (_Float16)W[(size_t)col * DIM + K];
}

// ---------------------------------------------------------------------------
// Kernel 1: partial kv = relu(k)^T * v and ksum per (b,h) per N-chunk.
// ---------------------------------------------------------------------------
__global__ __launch_bounds__(256) void kv_partial_kernel(
    const float* __restrict__ qkv, const unsigned char* __restrict__ mask,
    float* __restrict__ kvpart, float* __restrict__ kspart) {
    __shared__ float kv_s[HDIM * HDIM];
    __shared__ float ks_s[HDIM];

    const int tid  = threadIdx.x;
    const int lane = tid & 31, wave = tid >> 5;
    const int m    = lane & 15, half = lane >> 4;
    const int bh    = blockIdx.x / K1_CH;
    const int chunk = blockIdx.x % K1_CH;
    const int b = bh / NHEADS, h = bh % NHEADS;

    for (int i = tid; i < HDIM * HDIM; i += 256) kv_s[i] = 0.0f;
    if (tid < HDIM) ks_s[tid] = 0.0f;
    __syncthreads();

    v8f acc[4][4] = {};
    float ks[4] = {0.0f, 0.0f, 0.0f, 0.0f};
    const int nbase = chunk * K1_ROWS + wave * K1_WROWS;

    for (int it = 0; it < K1_ITERS; ++it) {
        const int n0 = nbase + it * 32;

        // Prefetch next iteration's k/v rows (one row per lane).
        {
            int npf = n0 + 32 + lane;
            npf = npf < SEQ ? npf : (SEQ - 1);
            const size_t prow = (size_t)(b * SEQ + npf) * (3 * DIM);
            __builtin_prefetch(&qkv[prow + DIM + h * HDIM], 0, 1);
            __builtin_prefetch(&qkv[prow + 2 * DIM + h * HDIM], 0, 1);
        }

        // B fragments: v tiles (32 x 16), masked. b[i] <-> row n0+a_kpos(i,half), col f0+m.
        v16h bf[4];
#pragma unroll
        for (int tf = 0; tf < 4; ++tf) {
            const int f0 = tf * 16;
#pragma unroll
            for (int i = 0; i < 16; ++i) {
                const int n = n0 + a_kpos(i, half);
                float v = qkv[(size_t)(b * SEQ + n) * (3 * DIM) + 2 * DIM + h * HDIM + f0 + m];
                if (!mask[b * SEQ + n]) v = 0.0f;
                bf[tf][i] = (_Float16)v;
            }
        }
        // A fragments: k^T tiles (16 x 32), relu+mask; also feed ksum.
#pragma unroll
        for (int td = 0; td < 4; ++td) {
            const int d0 = td * 16;
            v16h af;
#pragma unroll
            for (int i = 0; i < 16; ++i) {
                const int n = n0 + a_kpos(i, half);
                float v = qkv[(size_t)(b * SEQ + n) * (3 * DIM) + DIM + h * HDIM + d0 + m];
                v = fmaxf(v, 0.0f);
                if (!mask[b * SEQ + n]) v = 0.0f;
                af[i] = (_Float16)v;
                ks[td] += v;
            }
#pragma unroll
            for (int tf = 0; tf < 4; ++tf)
                acc[td][tf] = wmma_f16(af, bf[tf], acc[td][tf]);
        }
    }

    // combine lane m / m+16 halves of ksum (they cover disjoint K ranges)
    float ksc[4];
#pragma unroll
    for (int td = 0; td < 4; ++td) {
        float s = ks[td];
        s += __shfl_xor(s, 16, 32);
        ksc[td] = s;
    }

    // deterministic cross-wave reduction in LDS
    for (int w = 0; w < 8; ++w) {
        if (wave == w) {
#pragma unroll
            for (int td = 0; td < 4; ++td) {
#pragma unroll
                for (int tf = 0; tf < 4; ++tf)
#pragma unroll
                    for (int j = 0; j < 8; ++j) {
                        const int row = td * 16 + j + 8 * half;   // d
                        const int col = tf * 16 + m;              // f
                        kv_s[row * HDIM + col] += acc[td][tf][j];
                    }
                if (half == 0) ks_s[td * 16 + m] += ksc[td];
            }
        }
        __syncthreads();
    }

    float* kvp = kvpart + (size_t)blockIdx.x * (HDIM * HDIM);
    for (int i = tid; i < HDIM * HDIM; i += 256) kvp[i] = kv_s[i];
    if (tid < HDIM) kspart[blockIdx.x * HDIM + tid] = ks_s[tid];
}

// ---------------------------------------------------------------------------
// Kernel 2: reduce chunk partials -> kv (f32), ksum (f32). One block per (b,h).
// ---------------------------------------------------------------------------
__global__ __launch_bounds__(256) void kv_reduce_kernel(
    const float* __restrict__ kvpart, const float* __restrict__ kspart,
    float* __restrict__ kvf, float* __restrict__ ksumf) {
    const int bh = blockIdx.x;
    for (int i = threadIdx.x; i < HDIM * HDIM; i += 256) {
        float s = 0.0f;
#pragma unroll
        for (int c = 0; c < K1_CH; ++c)
            s += kvpart[((size_t)(bh * K1_CH + c)) * (HDIM * HDIM) + i];
        kvf[(size_t)bh * (HDIM * HDIM) + i] = s;
    }
    if (threadIdx.x < HDIM) {
        float s = 0.0f;
#pragma unroll
        for (int c = 0; c < K1_CH; ++c)
            s += kspart[(bh * K1_CH + c) * HDIM + threadIdx.x];
        ksumf[bh * HDIM + threadIdx.x] = s;
    }
}

// ---------------------------------------------------------------------------
// Kernel 3: att = (relu(q) @ kv) / (relu(q)·ksum + 1e-6), stored f16 row-major.
// kv state staged into LDS with gfx1250 async global->LDS copies when available.
// ---------------------------------------------------------------------------
__global__ __launch_bounds__(256) void attn_norm_kernel(
    const float* __restrict__ qkv, const float* __restrict__ kvf,
    const float* __restrict__ ksumf, _Float16* __restrict__ att16) {
    __shared__ _Float16 kv16_s[HDIM * HDIM];
    __shared__ float ks_s[HDIM];

    const int tid  = threadIdx.x;
    const int lane = tid & 31, wave = tid >> 5;
    const int m    = lane & 15, half = lane >> 4;
    const int bh    = blockIdx.x / K2_BLKS;
    const int chunk = blockIdx.x % K2_BLKS;
    const int b = bh / NHEADS, h = bh % NHEADS;

#if HAVE_ASYNC_LDS
    __shared__ float kvf_s[HDIM * HDIM];
    {
        // Async DMA of the 16KB kv state into LDS (b128 per lane, 4 deep).
        const uintptr_t g = (uintptr_t)(kvf + (size_t)bh * (HDIM * HDIM));
        const unsigned int l = (unsigned int)(uintptr_t)&kvf_s[0];
#pragma unroll
        for (int c = 0; c < 4; ++c) {
            const unsigned int byteoff = (unsigned int)(c * 256 + tid) * 16u;
            __builtin_amdgcn_global_load_async_to_lds_b128(
                (gas_v4i_ptr)(g + (uintptr_t)byteoff),
                (lds_v4i_ptr)(l + byteoff),
                0, 0);
        }
#if defined(__has_builtin) && __has_builtin(__builtin_amdgcn_s_wait_asynccnt)
        __builtin_amdgcn_s_wait_asynccnt(0);
#else
        asm volatile("s_wait_asynccnt 0" ::: "memory");
#endif
    }
    __syncthreads();
    for (int i = tid; i < HDIM * HDIM; i += 256)
        kv16_s[i] = (_Float16)kvf_s[i];
#else
    for (int i = tid; i < HDIM * HDIM; i += 256)
        kv16_s[i] = (_Float16)kvf[(size_t)bh * (HDIM * HDIM) + i];
#endif
    if (tid < HDIM) ks_s[tid] = ksumf[bh * HDIM + tid];
    __syncthreads();

    // Hoist B fragments of kv (8 total: 2 K-blocks x 4 f-tiles).
    v16h bf[2][4];
#pragma unroll
    for (int dblk = 0; dblk < 2; ++dblk)
#pragma unroll
        for (int tf = 0; tf < 4; ++tf)
#pragma unroll
            for (int i = 0; i < 16; ++i)
                bf[dblk][tf][i] = kv16_s[(dblk * 32 + a_kpos(i, half)) * HDIM + tf * 16 + m];

    // ksum gathered in A-fragment order for the per-lane denominator dot.
    float ksA[2][16];
#pragma unroll
    for (int dblk = 0; dblk < 2; ++dblk)
#pragma unroll
        for (int i = 0; i < 16; ++i)
            ksA[dblk][i] = ks_s[dblk * 32 + a_kpos(i, half)];

    for (int tt = 0; tt < 8; ++tt) {
        const int n0   = chunk * K2_ROWS + (wave * 8 + tt) * 16;
        const int nrow = n0 + m;  // this lane's A row
        const float* qrow = qkv + (size_t)(b * SEQ + nrow) * (3 * DIM) + h * HDIM;

        // Prefetch next tile's q rows (16 rows, both 128B halves via lane half).
        {
            int npf = n0 + 16 + m;
            npf = npf < SEQ ? npf : (SEQ - 1);
            __builtin_prefetch(qkv + (size_t)(b * SEQ + npf) * (3 * DIM) + h * HDIM + 32 * half, 0, 1);
        }

        v16h a0, a1;
        float dn = 0.0f;
#pragma unroll
        for (int i = 0; i < 16; ++i) {
            const int d = a_kpos(i, half);
            const float v0 = fmaxf(qrow[d], 0.0f);
            const float v1 = fmaxf(qrow[32 + d], 0.0f);
            a0[i] = (_Float16)v0;
            a1[i] = (_Float16)v1;
            dn += v0 * ksA[0][i] + v1 * ksA[1][i];
        }
        dn += __shfl_xor(dn, 16, 32);  // full denom for row (lane&15)

        v8f c[4] = {};
#pragma unroll
        for (int tf = 0; tf < 4; ++tf) {
            c[tf] = wmma_f16(a0, bf[0][tf], c[tf]);
            c[tf] = wmma_f16(a1, bf[1][tf], c[tf]);
        }

#pragma unroll
        for (int j = 0; j < 8; ++j) {
            const int row   = n0 + j + 8 * half;
            const float dr  = __shfl(dn, j + 8 * half, 32);
            const float rin = 1.0f / (dr + 1e-6f);
            _Float16* orow = att16 + (size_t)(b * SEQ + row) * DIM + h * HDIM;
#pragma unroll
            for (int tf = 0; tf < 4; ++tf)
                orow[tf * 16 + m] = (_Float16)(c[tf][j] * rin);
        }
    }
}

// ---------------------------------------------------------------------------
// Kernel 4: out = att16 @ W_out^T + b_out.  Each wave: 32x64 strip, K=1024.
// B fragments (wfrag, pre-permuted) are one 32B coalesced load per lane and
// reused across both 16-row tiles: 8 WMMA per 8 loads per K-step.
// ---------------------------------------------------------------------------
__global__ __launch_bounds__(256) void out_proj_kernel(
    const _Float16* __restrict__ att16, const _Float16* __restrict__ wfrag,
    const float* __restrict__ bias, float* __restrict__ out) {
    const int tid  = threadIdx.x;
    const int lane = tid & 31, wave = tid >> 5;
    const int m    = lane & 15, half = lane >> 4;

    const int mpair = blockIdx.x * 8 + wave;  // 0..511 (32-row tile pairs)
    const int m0    = mpair * 32;
    const int jt0   = blockIdx.y * 4;         // 4 j-tiles = 64 output cols

    v8f acc[2][4] = {};
    const v8h* ap0 = (const v8h*)(att16 + (size_t)(m0 + m) * DIM);
    const v8h* ap1 = (const v8h*)(att16 + (size_t)(m0 + 16 + m) * DIM);
    const v16h* wf = (const v16h*)wfrag;

    for (int kt = 0; kt < DIM / 32; ++kt) {
        // A fragments: two contiguous 16B loads per row tile from row-major att16.
        const v8h lo0 = ap0[kt * 4 + half];
        const v8h hi0 = ap0[kt * 4 + 2 + half];
        const v8h lo1 = ap1[kt * 4 + half];
        const v8h hi1 = ap1[kt * 4 + 2 + half];
        v16h a0, a1;
#pragma unroll
        for (int i = 0; i < 8; ++i) {
            a0[i] = lo0[i]; a0[8 + i] = hi0[i];
            a1[i] = lo1[i]; a1[8 + i] = hi1[i];
        }

#pragma unroll
        for (int tf = 0; tf < 4; ++tf) {
            const v16h bfr = wf[((size_t)(jt0 + tf) * 32 + kt) * 32 + lane];
            acc[0][tf] = wmma_f16(a0, bfr, acc[0][tf]);
            acc[1][tf] = wmma_f16(a1, bfr, acc[1][tf]);
        }
    }

#pragma unroll
    for (int r = 0; r < 2; ++r)
#pragma unroll
        for (int j = 0; j < 8; ++j) {
            const int row = m0 + r * 16 + j + 8 * half;
#pragma unroll
            for (int tf = 0; tf < 4; ++tf) {
                const int col = (jt0 + tf) * 16 + m;
                out[(size_t)row * DIM + col] = acc[r][tf][j] + bias[col];
            }
        }
}

// ---------------------------------------------------------------------------
extern "C" void kernel_launch(void* const* d_in, const int* in_sizes, int n_in,
                              void* d_out, int out_size, void* d_ws, size_t ws_size,
                              hipStream_t stream) {
    (void)in_sizes; (void)n_in; (void)out_size; (void)ws_size;
    const float*         qkv  = (const float*)d_in[1];
    const unsigned char* mask = (const unsigned char*)d_in[2];
    const float*         W    = (const float*)d_in[3];
    const float*         bias = (const float*)d_in[4];
    float*               out  = (float*)d_out;

    char* ws = (char*)d_ws;
    size_t off = 0;
    float* kvpart = (float*)(ws + off); off += (size_t)BATCH * NHEADS * K1_CH * HDIM * HDIM * sizeof(float);
    float* kspart = (float*)(ws + off); off += (size_t)BATCH * NHEADS * K1_CH * HDIM * sizeof(float);
    float* kvf    = (float*)(ws + off); off += (size_t)BATCH * NHEADS * HDIM * HDIM * sizeof(float);
    float* ksumf  = (float*)(ws + off); off += (size_t)BATCH * NHEADS * HDIM * sizeof(float);
    _Float16* wfrag = (_Float16*)(ws + off); off += (size_t)DIM * DIM * sizeof(_Float16);
    _Float16* att16 = (_Float16*)(ws + off); off += (size_t)NROWS * DIM * sizeof(_Float16);

    wfrag_kernel<<<DIM * DIM / 256, 256, 0, stream>>>(W, wfrag);
    kv_partial_kernel<<<BATCH * NHEADS * K1_CH, 256, 0, stream>>>(qkv, mask, kvpart, kspart);
    kv_reduce_kernel<<<BATCH * NHEADS, 256, 0, stream>>>(kvpart, kspart, kvf, ksumf);
    attn_norm_kernel<<<BATCH * NHEADS * K2_BLKS, 256, 0, stream>>>(qkv, kvf, ksumf, att16);
    out_proj_kernel<<<dim3(NROWS / 32 / 8, DIM / 64), 256, 0, stream>>>(att16, wfrag, bias, out);
}
